// CausalSelfAttention_82489141887299
// MI455X (gfx1250) — matrix operand order repaired and checked by reading, not compiled, not run on hardware
//
#include <hip/hip_runtime.h>

typedef __attribute__((ext_vector_type(16))) _Float16 v16h;
typedef __attribute__((ext_vector_type(8)))  _Float16 v8h;
typedef __attribute__((ext_vector_type(8)))  float    v8f;

namespace {
constexpr int Bc = 4, Tc = 2048, Dc = 1024, Hc = 16, HDc = 64;
constexpr int BM = 128, BN = 128, BK = 32, LDT = 40; // GEMM LDS stride (halves)
constexpr int PSTR = 72;                             // attention P stride (64 keys + pad)
}

__device__ __forceinline__ v8f wmma16(v16h a, v16h b, v8f c) {
  return __builtin_amdgcn_wmma_f32_16x16x32_f16(false, a, false, b, (short)0, c, false, false);
}

__device__ __forceinline__ v16h joinh(v8h lo, v8h hi) {
  union { v16h v; v8h h[2]; } u; u.h[0] = lo; u.h[1] = hi; return u.v;
}

// ---- DPP 16-lane reductions (stay within each half-wave row) ----
template <int CTRL>
__device__ __forceinline__ float dppmov(float x) {
  int xi = __builtin_bit_cast(int, x);
  return __builtin_bit_cast(float,
      __builtin_amdgcn_update_dpp(xi, xi, CTRL, 0xf, 0xf, true));
}
__device__ __forceinline__ float red_max16(float x) {
  x = fmaxf(x, dppmov<0xB1>(x));   // quad_perm [1,0,3,2]  (xor 1)
  x = fmaxf(x, dppmov<0x4E>(x));   // quad_perm [2,3,0,1]  (xor 2)
  x = fmaxf(x, dppmov<0x141>(x));  // row_half_mirror      (xor 4 eff.)
  x = fmaxf(x, dppmov<0x140>(x));  // row_mirror           (xor 8 eff.)
  return x;
}
__device__ __forceinline__ float red_sum16(float x) {
  x += dppmov<0xB1>(x);
  x += dppmov<0x4E>(x);
  x += dppmov<0x141>(x);
  x += dppmov<0x140>(x);
  return x;
}

// gfx1250 async copy: 16B per lane, global -> LDS, tracked by ASYNCcnt.
__device__ __forceinline__ void async_cp16(_Float16* lds, const _Float16* g) {
  asm volatile("global_load_async_to_lds_b128 %0, %1, off"
               :: "v"((unsigned)(uintptr_t)lds), "v"((unsigned long long)(uintptr_t)g)
               : "memory");
}
__device__ __forceinline__ void wait_async4() {
  asm volatile("s_wait_asynccnt 4" ::: "memory");
}
__device__ __forceinline__ void wait_async0() {
  asm volatile("s_wait_asynccnt 0" ::: "memory");
}

// ---------------- precision-convert / transpose helpers ----------------
__global__ void cvt_f32_f16(const float* __restrict__ in, _Float16* __restrict__ o, int n) {
  int i = blockIdx.x * blockDim.x + threadIdx.x;
  if (i < n) o[i] = (_Float16)in[i];
}

__global__ void transpose_f32_f16(const float* __restrict__ w, _Float16* __restrict__ wt,
                                  int K, int N) {
  int i = blockIdx.x * blockDim.x + threadIdx.x;
  if (i < K * N) {
    int n = i / K, k = i - n * K;
    wt[i] = (_Float16)w[(size_t)k * N + n];
  }
}

// ---------------- WMMA GEMM: C[M,N] = A[M,K] x Bw[N,K]^T ----------------
template <int MODE>
__global__ __launch_bounds__(256) void gemm_wmma(const _Float16* __restrict__ A,
                                                 const _Float16* __restrict__ Bw,
                                                 _Float16* __restrict__ qh,
                                                 _Float16* __restrict__ kh,
                                                 _Float16* __restrict__ vt,
                                                 float* __restrict__ outp) {
  constexpr int K = Dc;
  constexpr int TILEH = BM * LDT;
  __shared__ _Float16 As[2][TILEH];
  __shared__ _Float16 Bs[2][TILEH];
  const int tid  = threadIdx.x;
  const int lane = tid & 31;
  const int wave = tid >> 5;
  const int hlf  = lane >> 4;
  const int l16  = lane & 15;
  const int m0 = blockIdx.y * BM;
  const int n0 = blockIdx.x * BN;
  const int wm = (wave & 1) * 64;
  const int wn = (wave >> 1) * 32;

  v8f zero = {};
  v8f acc[4][2];
#pragma unroll
  for (int im = 0; im < 4; ++im)
#pragma unroll
    for (int in = 0; in < 2; ++in) acc[im][in] = zero;

  auto stage = [&](int buf, int k0) {
#pragma unroll
    for (int i = 0; i < 2; ++i) {
      int c   = i * 256 + tid;
      int row = c >> 2;
      int kc  = (c & 3) * 8;
      async_cp16(&As[buf][row * LDT + kc], &A[(size_t)(m0 + row) * K + k0 + kc]);
      async_cp16(&Bs[buf][row * LDT + kc], &Bw[(size_t)(n0 + row) * K + k0 + kc]);
    }
  };

  stage(0, 0);
  const int nk = K / BK;
  for (int it = 0; it < nk; ++it) {
    int cur = it & 1;
    if (it + 1 < nk) {
      stage(cur ^ 1, (it + 1) * BK);
      wait_async4();
    } else {
      wait_async0();
    }
    __syncthreads();

    v16h a[4], b[2];
#pragma unroll
    for (int im = 0; im < 4; ++im) {
      int r = wm + im * 16 + l16;
      v8h lo = *(const v8h*)&As[cur][r * LDT + 8 * hlf];
      v8h hi = *(const v8h*)&As[cur][r * LDT + 16 + 8 * hlf];
      a[im] = joinh(lo, hi);
    }
#pragma unroll
    for (int in = 0; in < 2; ++in) {
      int r = wn + in * 16 + l16;
      v8h lo = *(const v8h*)&Bs[cur][r * LDT + 8 * hlf];
      v8h hi = *(const v8h*)&Bs[cur][r * LDT + 16 + 8 * hlf];
      b[in] = joinh(lo, hi);
    }
#pragma unroll
    for (int im = 0; im < 4; ++im)
#pragma unroll
      for (int in = 0; in < 2; ++in)
        acc[im][in] = wmma16(a[im], b[in], acc[im][in]);
    __syncthreads();
  }

#pragma unroll
  for (int im = 0; im < 4; ++im)
#pragma unroll
    for (int in = 0; in < 2; ++in) {
      int n = n0 + wn + in * 16 + l16;
#pragma unroll
      for (int v = 0; v < 8; ++v) {
        int   m   = m0 + wm + im * 16 + v + 8 * hlf;
        float val = acc[im][in][v];
        if constexpr (MODE == 0) {
          int bb = m >> 11, t = m & (Tc - 1);
          if (n < Dc) {
            int h = n >> 6, d = n & 63;
            qh[(((size_t)bb * Hc + h) * Tc + t) * HDc + d] = (_Float16)(val * 0.125f);
          } else if (n < 2 * Dc) {
            int nn = n - Dc; int h = nn >> 6, d = nn & 63;
            kh[(((size_t)bb * Hc + h) * Tc + t) * HDc + d] = (_Float16)val;
          } else {
            int nn = n - 2 * Dc; int h = nn >> 6, d = nn & 63;
            vt[(((size_t)bb * Hc + h) * HDc + d) * Tc + t] = (_Float16)val;  // V^T
          }
        } else {
          outp[(size_t)m * Dc + n] = val;
        }
      }
    }
}

// ---------------- flash attention: 64-key blocks ----------------
template <bool MASK>
__device__ __forceinline__ void attn_kblock64(const _Float16* __restrict__ Kp,
                                              const _Float16* __restrict__ Vp,
                                              _Float16* __restrict__ P,
                                              const v16h* qa, v8f* o,
                                              float* mrow, float* lrow,
                                              int qbase, int kb, int hlf, int l16) {
  v8f zero = {};
  v8f s[4];
#pragma unroll
  for (int t = 0; t < 4; ++t) {                  // S = Q K^T, 4 key sub-tiles
    int key = kb + 16 * t + l16;
    v8f sa = zero;
#pragma unroll
    for (int st = 0; st < 2; ++st) {
      v8h lo = *(const v8h*)&Kp[(size_t)key * HDc + 32 * st + 8 * hlf];
      v8h hi = *(const v8h*)&Kp[(size_t)key * HDc + 32 * st + 16 + 8 * hlf];
      sa = wmma16(qa[st], joinh(lo, hi), sa);
    }
    s[t] = sa;
  }
  if constexpr (MASK) {
#pragma unroll
    for (int t = 0; t < 4; ++t) {
      int key = kb + 16 * t + l16;
#pragma unroll
      for (int v = 0; v < 8; ++v)
        if (key > qbase + v + 8 * hlf) s[t][v] = -3.0e38f;
    }
  }
  float mnew[8], alpha[8];
#pragma unroll
  for (int v = 0; v < 8; ++v) {
    float t = fmaxf(fmaxf(s[0][v], s[1][v]), fmaxf(s[2][v], s[3][v]));
    t = red_max16(t);
    mnew[v]  = fmaxf(mrow[v], t);
    alpha[v] = __expf(mrow[v] - mnew[v]);
    mrow[v]  = mnew[v];
  }
#pragma unroll
  for (int t = 0; t < 4; ++t)
#pragma unroll
    for (int v = 0; v < 8; ++v) {
      float p = __expf(s[t][v] - mnew[v]);
      s[t][v] = p;
      P[(v + 8 * hlf) * PSTR + 16 * t + l16] = (_Float16)p;  // C-layout -> [row][key]
    }
#pragma unroll
  for (int v = 0; v < 8; ++v) {
    float r = (s[0][v] + s[1][v]) + (s[2][v] + s[3][v]);
    r = red_sum16(r);
    lrow[v] = lrow[v] * alpha[v] + r;
  }
#pragma unroll
  for (int nt = 0; nt < 4; ++nt)
#pragma unroll
    for (int v = 0; v < 8; ++v) o[nt][v] *= alpha[v];

  asm volatile("s_wait_dscnt 0" ::: "memory");   // wave-local LDS store->load

#pragma unroll
  for (int hb = 0; hb < 2; ++hb) {               // two 32-key halves: O += P * V
    v8h lo = *(const v8h*)&P[l16 * PSTR + 32 * hb + 8 * hlf];
    v8h hi = *(const v8h*)&P[l16 * PSTR + 32 * hb + 16 + 8 * hlf];
    v16h pa = joinh(lo, hi);
#pragma unroll
    for (int nt = 0; nt < 4; ++nt) {
      int d = 16 * nt + l16;
      v8h vlo = *(const v8h*)&Vp[(size_t)d * Tc + kb + 32 * hb + 8 * hlf];
      v8h vhi = *(const v8h*)&Vp[(size_t)d * Tc + kb + 32 * hb + 16 + 8 * hlf];
      o[nt] = wmma16(pa, joinh(vlo, vhi), o[nt]);
    }
  }
  asm volatile("" ::: "memory");
}

__global__ __launch_bounds__(256) void flash_attn(const _Float16* __restrict__ qh,
                                                  const _Float16* __restrict__ kh,
                                                  const _Float16* __restrict__ vt,
                                                  _Float16* __restrict__ yh) {
  __shared__ _Float16 Pl[8][16 * PSTR];   // per-wave P scratch
  const int tid  = threadIdx.x;
  const int lane = tid & 31;
  const int wave = tid >> 5;
  const int hlf  = lane >> 4;
  const int l16  = lane & 15;
  _Float16* P = &Pl[wave][0];

  int gtile = blockIdx.x * 8 + wave;          // 0 .. B*H*(T/16)-1
  int qt    = gtile & (Tc / 16 - 1);
  int bh    = gtile >> 7;
  int qbase = qt * 16;
  const _Float16* Q  = qh + (size_t)bh * Tc * HDc;
  const _Float16* Kp = kh + (size_t)bh * Tc * HDc;
  const _Float16* Vp = vt + (size_t)bh * HDc * Tc;

  v16h qa[2];
  {
    int qrow = qbase + l16;
#pragma unroll
    for (int s = 0; s < 2; ++s) {
      v8h lo = *(const v8h*)&Q[(size_t)qrow * HDc + 32 * s + 8 * hlf];
      v8h hi = *(const v8h*)&Q[(size_t)qrow * HDc + 32 * s + 16 + 8 * hlf];
      qa[s] = joinh(lo, hi);
    }
  }

  v8f zero = {};
  v8f o[4];
#pragma unroll
  for (int i = 0; i < 4; ++i) o[i] = zero;
  float mrow[8], lrow[8];
#pragma unroll
  for (int v = 0; v < 8; ++v) { mrow[v] = -3.0e38f; lrow[v] = 0.0f; }

  // full 64-key blocks below the diagonal, then one masked diagonal block
  const int kfull = (qbase + 1) >> 6;
  for (int i = 0; i < kfull; ++i)
    attn_kblock64<false>(Kp, Vp, P, qa, o, mrow, lrow, qbase, 64 * i, hlf, l16);
  attn_kblock64<true>(Kp, Vp, P, qa, o, mrow, lrow, qbase, 64 * kfull, hlf, l16);

  int bb = bh >> 4, h = bh & 15;
#pragma unroll
  for (int nt = 0; nt < 4; ++nt)
#pragma unroll
    for (int v = 0; v < 8; ++v) {
      int tok = qbase + v + 8 * hlf;
      float val = o[nt][v] / lrow[v];
      yh[((size_t)bb * Tc + tok) * Dc + h * 64 + nt * 16 + l16] = (_Float16)val;
    }
}

// ---------------- host-side orchestration ----------------
extern "C" void kernel_launch(void* const* d_in, const int* in_sizes, int n_in,
                              void* d_out, int out_size, void* d_ws, size_t ws_size,
                              hipStream_t stream) {
  const float* x      = (const float*)d_in[0];
  const float* w_attn = (const float*)d_in[1];
  const float* w_proj = (const float*)d_in[2];
  float* out = (float*)d_out;

  _Float16* ws = (_Float16*)d_ws;
  size_t off = 0;
  _Float16* xh     = ws + off; off += (size_t)Bc * Tc * Dc;
  _Float16* wqkvt  = ws + off; off += (size_t)Dc * 3 * Dc;
  _Float16* wprojt = ws + off; off += (size_t)Dc * Dc;
  _Float16* qh     = ws + off; off += (size_t)Bc * Tc * Dc;
  _Float16* kh     = ws + off; off += (size_t)Bc * Tc * Dc;
  _Float16* vt     = ws + off; off += (size_t)Bc * Tc * Dc;
  _Float16* yh     = ws + off; off += (size_t)Bc * Tc * Dc;

  const int nx = Bc * Tc * Dc;
  cvt_f32_f16<<<nx / 256, 256, 0, stream>>>(x, xh, nx);
  transpose_f32_f16<<<(Dc * 3 * Dc) / 256, 256, 0, stream>>>(w_attn, wqkvt, Dc, 3 * Dc);
  transpose_f32_f16<<<(Dc * Dc) / 256, 256, 0, stream>>>(w_proj, wprojt, Dc, Dc);

  gemm_wmma<0><<<dim3(3 * Dc / BN, Bc * Tc / BM), 256, 0, stream>>>(
      xh, wqkvt, qh, kh, vt, nullptr);

  flash_attn<<<(Bc * Hc * (Tc / 16)) / 8, 256, 0, stream>>>(qh, kh, vt, yh);

  gemm_wmma<1><<<dim3(Dc / BN, Bc * Tc / BM), 256, 0, stream>>>(
      yh, wprojt, nullptr, nullptr, nullptr, out);
}
